// LambdaModule_46772193854059
// MI455X (gfx1250) — compile-verified, matrix operand order checked by reading, not run on hardware
//
#include <hip/hip_runtime.h>

typedef __bf16 v16bf __attribute__((ext_vector_type(16)));
typedef __bf16 v8bf  __attribute__((ext_vector_type(8)));
typedef __bf16 v4bf  __attribute__((ext_vector_type(4)));
typedef float  v8f   __attribute__((ext_vector_type(8)));

constexpr int B = 4, H = 8, N = 2048, K = 64, M = 2048, V = 64;
constexpr int MC = 64;   // M rows staged per chunk
constexpr int MS = 72;   // padded LDS row stride (halves): 144B rows keep 16B
                         // alignment for b128 frag loads, spread store banks

// ---------------------------------------------------------------------------
// Kernel 1: P[b,m,:] = softmax(keys[b,m,:]) over k, stored bf16 in workspace.
// ---------------------------------------------------------------------------
__global__ __launch_bounds__(256)
void softmax_rows_kernel(const float* __restrict__ keys, __bf16* __restrict__ P) {
  const int row = blockIdx.x * blockDim.x + threadIdx.x;   // over B*M = 8192
  const float* r = keys + (size_t)row * K;
  float mx = -3.4e38f;
  #pragma unroll 8
  for (int k = 0; k < K; ++k) mx = fmaxf(mx, r[k]);
  float s = 0.f;
  #pragma unroll 8
  for (int k = 0; k < K; ++k) s += __expf(r[k] - mx);
  const float inv = 1.f / s;
  __bf16* o = P + (size_t)row * K;
  #pragma unroll 8
  for (int k = 0; k < K; ++k) o[k] = (__bf16)(__expf(r[k] - mx) * inv);
}

// Assemble a 16-half WMMA fragment from two 16B LDS loads:
// halves [0..7] at p, halves [8..15] at p+16 (matches 16-bit A/B lane layout).
__device__ __forceinline__ v16bf frag_ld(const __bf16* p) {
  v8bf lo = *(const v8bf*)p;
  v8bf hi = *(const v8bf*)(p + 16);
  return __builtin_shufflevector(lo, hi, 0, 1, 2, 3, 4, 5, 6, 7,
                                 8, 9, 10, 11, 12, 13, 14, 15);
}

// ---------------------------------------------------------------------------
// Kernel 2: per (b,n): lambda = (P[b] + E[n])^T @ V[b]  (64x64, f32 acc, bf16
// operands, WMMA over M=2048), then out[b,n,h,:] = Q[b,h,n,:] @ lambda.
// ---------------------------------------------------------------------------
__global__ __launch_bounds__(256)
void lambda_fused_kernel(const float* __restrict__ Q,
                         const float* __restrict__ Vg,
                         const float* __restrict__ E,
                         const __bf16* __restrict__ P,
                         float* __restrict__ out) {
  __shared__ __bf16 At[2][K][MS];   // A^T: [k][m], combined P+E, bf16
  __shared__ __bf16 Bt[2][V][MS];   // V^T: [v][m], bf16
  __shared__ float  lam[K][V];      // 64x64 lambda
  __shared__ float  qs[H][K];       // query slice for this (b,n)

  const int bid  = blockIdx.x;      // n*B + b  -> 4 adjacent blocks share E[n]
  const int n    = bid >> 2;
  const int b    = bid & 3;
  const int t    = threadIdx.x;
  const int lane = t & 31;
  const int wave = t >> 5;
  const int half = lane >> 4;       // 0/1: lane group selects m sub-range
  const int l15  = lane & 15;
  const int ktile = wave & 3;       // 4 k-tiles of 16
  const int vgrp  = wave >> 2;      // 2 v-groups of 32 (2 tiles each)
  const int arow  = ktile * 16 + l15;
  const int vcol  = vgrp * 32 + l15;

  // Stage Q[b,:,n,:] (8x64 f32) into LDS.
  {
    int o = t;
    #pragma unroll
    for (int i = 0; i < 2; ++i, o += 256) {
      const int h = o >> 6, k = o & 63;
      qs[h][k] = Q[(((size_t)b * H + h) * N + n) * K + k];
    }
  }

  const float*  Eb = E  + (size_t)n * M * K;
  const float*  Vb = Vg + (size_t)b * M * V;
  const __bf16* Pb = P  + (size_t)b * M * K;

  // Stage one MC x 64 chunk of (P+E) and V, transposed+converted, into buf.
  auto stage = [&](int c, int buf) {
    const int m0 = c * MC;
    #pragma unroll
    for (int i = 0; i < 4; ++i) {
      const int u  = t + i * 256;   // 0..1023 over (m_local, kc)
      const int kc = u & 15;        // float4 column
      const int ml = u >> 4;        // 0..63 m row
      const size_t mrow = (size_t)(m0 + ml);
      const float4 e4 = *(const float4*)(Eb + mrow * K + kc * 4);
      const v4bf   p4 = *(const v4bf*)(Pb + mrow * K + kc * 4);
      const float4 v4 = *(const float4*)(Vb + mrow * V + kc * 4);
      const int kb = kc * 4;
      At[buf][kb + 0][ml] = (__bf16)(e4.x + (float)p4[0]);
      At[buf][kb + 1][ml] = (__bf16)(e4.y + (float)p4[1]);
      At[buf][kb + 2][ml] = (__bf16)(e4.z + (float)p4[2]);
      At[buf][kb + 3][ml] = (__bf16)(e4.w + (float)p4[3]);
      Bt[buf][kb + 0][ml] = (__bf16)v4.x;
      Bt[buf][kb + 1][ml] = (__bf16)v4.y;
      Bt[buf][kb + 2][ml] = (__bf16)v4.z;
      Bt[buf][kb + 3][ml] = (__bf16)v4.w;
      if (m0 + MC < M)  // pull next E chunk toward L2/L0
        __builtin_prefetch(Eb + (mrow + MC) * K + kc * 4, 0, 1);
    }
  };

  v8f acc0 = {};
  v8f acc1 = {};

  stage(0, 0);
  __syncthreads();

  constexpr int NCHUNK = M / MC;    // 32
  for (int c = 0; c < NCHUNK; ++c) {
    const int cur = c & 1;
    if (c + 1 < NCHUNK) stage(c + 1, 1 - cur);
    #pragma unroll
    for (int s = 0; s < 2; ++s) {   // two K=32 WMMA steps per chunk
      const int mb = s * 32 + half * 8;
      const v16bf a  = frag_ld(&At[cur][arow][mb]);
      const v16bf b0 = frag_ld(&Bt[cur][vcol][mb]);
      acc0 = __builtin_amdgcn_wmma_f32_16x16x32_bf16(
          false, a, false, b0, (short)0, acc0, false, false);
      const v16bf b1 = frag_ld(&Bt[cur][vcol + 16][mb]);
      acc1 = __builtin_amdgcn_wmma_f32_16x16x32_bf16(
          false, a, false, b1, (short)0, acc1, false, false);
    }
    __syncthreads();
  }

  // Spill accumulators to LDS per the 16x16 f32 C layout:
  // lane group half -> M = r + 8*half, N = l15.
  #pragma unroll
  for (int r = 0; r < 8; ++r) {
    lam[ktile * 16 + half * 8 + r][vgrp * 32 + l15]      = acc0[r];
    lam[ktile * 16 + half * 8 + r][vgrp * 32 + 16 + l15] = acc1[r];
  }
  __syncthreads();

  // Epilogue: out[b,n,h,v] = sum_k qs[h][k] * lam[k][v]  (content+position).
  int o = t;
  #pragma unroll
  for (int i = 0; i < 2; ++i, o += 256) {
    const int h = o >> 6, v = o & 63;
    float s = 0.f;
    #pragma unroll
    for (int kk = 0; kk < K; ++kk) s = fmaf(qs[h][kk], lam[kk][v], s);
    out[(((size_t)b * N + n) * H + h) * V + v] = s;
  }
}

// ---------------------------------------------------------------------------
extern "C" void kernel_launch(void* const* d_in, const int* in_sizes, int n_in,
                              void* d_out, int out_size, void* d_ws, size_t ws_size,
                              hipStream_t stream) {
  const float* queries   = (const float*)d_in[0];  // [B,H,N,K]
  const float* keys      = (const float*)d_in[1];  // [B,M,K]
  const float* values    = (const float*)d_in[2];  // [B,M,V]
  const float* embedding = (const float*)d_in[3];  // [N,M,K]
  float* out = (float*)d_out;                      // [B,N,H*V] flat
  __bf16* P = (__bf16*)d_ws;                       // [B,M,K] bf16 (1 MB)

  softmax_rows_kernel<<<(B * M) / 256, 256, 0, stream>>>(keys, P);
  lambda_fused_kernel<<<N * B, 256, 0, stream>>>(queries, values, embedding, P, out);
}
